// AttentionalFactorizationMachine_13073880449133
// MI455X (gfx1250) — matrix operand order repaired and checked by reading, not compiled, not run on hardware
//
#include <hip/hip_runtime.h>
#include <hip/hip_bf16.h>
#include <math.h>

#define NF        32
#define CARDN     10000
#define DIMD      64
#define NPAIR     496          // 32*31/2
#define NTILE     31           // 496/16
#define BLOCK     256
#define CHUNK_T   8            // M-tiles per chunk (== waves per block)
#define CHUNK_P   128          // pairs per chunk
#define NBATCH    4096

#define FAC_STRIDE   68        // floats  (bank-spread padding; 272B rows, 16B aligned)
#define INT_STRIDE   72        // bf16    (144B rows, 16B aligned)
#define W1T_STRIDE   72        // bf16

#if __has_builtin(__builtin_amdgcn_global_load_async_to_lds_b128) && \
    __has_builtin(__builtin_amdgcn_s_wait_asynccnt)
#define HAVE_ASYNC_LDS 1
#endif

typedef __attribute__((ext_vector_type(16))) __bf16 bf16x16;
typedef __attribute__((ext_vector_type(8)))  float  f32x8;
typedef int v4i __attribute__((vector_size(16)));
typedef __attribute__((address_space(1))) v4i* gptr_v4i;   // global (AS1)
typedef __attribute__((address_space(3))) v4i* lptr_v4i;   // LDS (AS3)

union FragU {
    bf16x16 v;
    uint4   q[2];
};

struct __align__(16) SmemT {
    float   factors[NF * FAC_STRIDE];        // 8704 B
    __bf16  inter[CHUNK_P * INT_STRIDE];     // 18432 B
    __bf16  w1t[DIMD * W1T_STRIDE];          // 9216 B
    float   logits[NPAIR];                   // 1984 B
    float   isum[NPAIR];                     // 1984 B
    float   b1s[DIMD];
    float   w2s[DIMD];
    int     xrow[NF];
};

// VALU-only cross-lane add via DPP16 (keeps the WMMA epilogue off the LDS path)
template <int CTRL>
__device__ __forceinline__ float dpp_addf(float v) {
    int t = __builtin_amdgcn_update_dpp(0, __float_as_int(v), CTRL, 0xf, 0xf, true);
    return v + __int_as_float(t);
}
#define DPP_QUAD_XOR1 0xB1     // quad_perm [1,0,3,2]
#define DPP_ROW_ROR1  0x121
#define DPP_ROW_ROR2  0x122
#define DPP_ROW_ROR4  0x124
#define DPP_ROW_ROR8  0x128

__device__ __forceinline__ void pair_ij(int p, int& i, int& j) {
    int ii = 0, rem = p;
    while (rem >= (NF - 1 - ii)) { rem -= (NF - 1 - ii); ++ii; }
    i = ii;
    j = ii + 1 + rem;
}

__global__ __launch_bounds__(BLOCK)
void afm_fused_kernel(const int*   __restrict__ x,
                      const float* __restrict__ emb,
                      const float* __restrict__ lin,
                      const float* __restrict__ lin_bias,
                      const float* __restrict__ W1,
                      const float* __restrict__ b1,
                      const float* __restrict__ W2,
                      const float* __restrict__ b2,
                      float*       __restrict__ out)
{
    __shared__ SmemT s;
    const int tid  = threadIdx.x;
    const int bidx = blockIdx.x;
    const int lane = tid & 31;
    const int wave = tid >> 5;

    // ---- phase 1: stage x-row, W1^T (bf16), b1, W2 -------------------------
    if (tid < NF) s.xrow[tid] = x[bidx * NF + tid];
    if (tid < DIMD) {
        s.b1s[tid] = b1[tid];
        s.w2s[tid] = W2[tid];
    }
    for (int i = tid; i < DIMD * DIMD; i += BLOCK) {
        int n = i & 63, k = i >> 6;                 // coalesced read of W1[k][n]
        s.w1t[n * W1T_STRIDE + k] = (__bf16)W1[k * DIMD + n];
    }
    __syncthreads();

    // ---- phase 2: gather embedding rows (async global->LDS if available) ---
#ifdef HAVE_ASYNC_LDS
    // 32 rows x 256B = 512 x b128 chunks; exactly 2 per thread -> EXEC all-1s
    for (int i = tid; i < NF * 16; i += BLOCK) {
        int f = i >> 4, c = i & 15;
        const float* gsrc = &emb[((size_t)f * CARDN + (size_t)s.xrow[f]) * DIMD + c * 4];
        float* ldst = &s.factors[f * FAC_STRIDE + c * 4];
        __builtin_amdgcn_global_load_async_to_lds_b128(
            (gptr_v4i)(void*)gsrc,
            (lptr_v4i)(void*)ldst,
            0, 0);
    }
    __builtin_amdgcn_s_wait_asynccnt(0);
#else
    for (int i = tid; i < NF * DIMD; i += BLOCK) {
        int f = i >> 6, d = i & 63;
        s.factors[f * FAC_STRIDE + d] =
            emb[((size_t)f * CARDN + (size_t)s.xrow[f]) * DIMD + d];
    }
#endif

    const int ncol = lane & 15;     // N within 16-wide tile / M-row within tile
    const int hilo = lane >> 4;     // lane half selects K sub-range
    const float b2v = b2[0];

    // Per-lane b1/W2 values for the 4 N-tiles this lane's column covers
    float b1v[4], w2v[4];
    #pragma unroll
    for (int nt = 0; nt < 4; ++nt) {
        b1v[nt] = s.b1s[nt * 16 + ncol];
        w2v[nt] = s.w2s[nt * 16 + ncol];
    }

    // B fragments: W1^T rows (contiguous-k) -> 32x16 bf16 operand layout
    FragU bf[2][4];
    #pragma unroll
    for (int kk = 0; kk < 2; ++kk) {
        #pragma unroll
        for (int nt = 0; nt < 4; ++nt) {
            const __bf16* wp =
                &s.w1t[(nt * 16 + ncol) * W1T_STRIDE + kk * 32 + hilo * 16];
            bf[kk][nt].q[0] = *(const uint4*)(wp);
            bf[kk][nt].q[1] = *(const uint4*)(wp + 8);
        }
    }
    __syncthreads();

    // ---- phase 3: chunked interactions + WMMA GEMM -------------------------
    for (int ch = 0; ch < 4; ++ch) {
        const int plo = ch * CHUNK_P;

        // build bf16 interactions for this chunk (+ per-pair sum)
        {
            int p  = plo + (tid >> 1);
            int dh = (tid & 1) * 32;
            float acc = 0.f;
            if (p < NPAIR) {
                int fi, fj;
                pair_ij(p, fi, fj);
                const float* fa = &s.factors[fi * FAC_STRIDE + dh];
                const float* fb = &s.factors[fj * FAC_STRIDE + dh];
                __bf16* dst = &s.inter[(p - plo) * INT_STRIDE + dh];
                #pragma unroll
                for (int d = 0; d < 32; ++d) {
                    float pr = fa[d] * fb[d];
                    acc += pr;
                    dst[d] = (__bf16)pr;
                }
            }
            acc = dpp_addf<DPP_QUAD_XOR1>(acc);      // combine the two d-halves
            if (((tid & 1) == 0) && p < NPAIR) s.isum[p] = acc;
        }
        __syncthreads();

        // one 16x64 output tile per wave
        const int mt = ch * CHUNK_T + wave;
        if (mt < NTILE) {
            f32x8 c0 = {}, c1 = {}, c2 = {}, c3 = {};
            #pragma unroll
            for (int kk = 0; kk < 2; ++kk) {
                // A fragment: 16x32 bf16, row = chunk-local pair row
                FragU a;
                const __bf16* rp =
                &s.inter[(wave * 16 + ncol) * INT_STRIDE + kk * 32 + hilo * 16];
            a.q[0] = *(const uint4*)(rp);
            a.q[1] = *(const uint4*)(rp + 8);

                c0 = __builtin_amdgcn_wmma_f32_16x16x32_bf16(
                        false, a.v, false, bf[kk][0].v, (short)0, c0, false, false);
                c1 = __builtin_amdgcn_wmma_f32_16x16x32_bf16(
                        false, a.v, false, bf[kk][1].v, (short)0, c1, false, false);
                c2 = __builtin_amdgcn_wmma_f32_16x16x32_bf16(
                        false, a.v, false, bf[kk][2].v, (short)0, c2, false, false);
                c3 = __builtin_amdgcn_wmma_f32_16x16x32_bf16(
                        false, a.v, false, bf[kk][3].v, (short)0, c3, false, false);
            }
            // epilogue: relu(+b1), dot with W2 over all 64 N,
            // 16-lane rotate-reduce entirely in VALU (DPP), no LDS traffic
            #pragma unroll
            for (int r = 0; r < 8; ++r) {
                float sacc = fmaxf(c0[r] + b1v[0], 0.f) * w2v[0]
                           + fmaxf(c1[r] + b1v[1], 0.f) * w2v[1]
                           + fmaxf(c2[r] + b1v[2], 0.f) * w2v[2]
                           + fmaxf(c3[r] + b1v[3], 0.f) * w2v[3];
                sacc = dpp_addf<DPP_ROW_ROR1>(sacc);
                sacc = dpp_addf<DPP_ROW_ROR2>(sacc);
                sacc = dpp_addf<DPP_ROW_ROR4>(sacc);
                sacc = dpp_addf<DPP_ROW_ROR8>(sacc);
                if (ncol == 0)
                    s.logits[mt * 16 + hilo * 8 + r] = sacc + b2v;
            }
        }
        __syncthreads();
    }

    // ---- phase 4: softmax over 496 pairs, weighted sum, linear term --------
    if (wave == 0) {
        float m = -3.402823e38f;
        for (int p = lane; p < NPAIR; p += 32)
            m = fmaxf(m, s.logits[p]);
        #pragma unroll
        for (int o = 16; o > 0; o >>= 1)
            m = fmaxf(m, __shfl_xor(m, o, 32));

        float se = 0.f, sa = 0.f;
        for (int p = lane; p < NPAIR; p += 32) {
            float e = __expf(s.logits[p] - m);
            se += e;
            sa += e * s.isum[p];
        }
        float lv = lin[(size_t)lane * CARDN + (size_t)s.xrow[lane]];
        #pragma unroll
        for (int o = 16; o > 0; o >>= 1) {
            se += __shfl_xor(se, o, 32);
            sa += __shfl_xor(sa, o, 32);
            lv += __shfl_xor(lv, o, 32);
        }
        if (lane == 0)
            out[bidx] = lv + lin_bias[0] + sa / se;
    }
}

extern "C" void kernel_launch(void* const* d_in, const int* in_sizes, int n_in,
                              void* d_out, int out_size, void* d_ws, size_t ws_size,
                              hipStream_t stream) {
    const int*   x    = (const int*)  d_in[0];
    const float* emb  = (const float*)d_in[1];
    const float* lin  = (const float*)d_in[2];
    const float* lb   = (const float*)d_in[3];
    const float* W1   = (const float*)d_in[4];
    const float* b1   = (const float*)d_in[5];
    const float* W2   = (const float*)d_in[6];
    const float* b2   = (const float*)d_in[7];
    float* out = (float*)d_out;

    afm_fused_kernel<<<NBATCH, BLOCK, 0, stream>>>(
        x, emb, lin, lb, W1, b1, W2, b2, out);
}
